// Unet_Both_61838939128041
// MI455X (gfx1250) — matrix operand order repaired and checked by reading, not compiled, not run on hardware
//
#include <hip/hip_runtime.h>
#include <hip/hip_bf16.h>
#include <stdint.h>

typedef __bf16 bf16;
typedef __bf16 v16bf __attribute__((ext_vector_type(16)));
typedef float  v8f   __attribute__((ext_vector_type(8)));

static inline int ceil_div(int a, int b) { return (a + b - 1) / b; }

// async global->LDS b128: VDST = LDS byte address (low 32 bits of generic
// shared pointer), VADDR = 64-bit global address, GV mode (no SADDR).
#define ASYNC_LD_B128(ldsa, ga)                                              \
  asm volatile("global_load_async_to_lds_b128 %0, %1, off" ::"v"(ldsa),      \
               "v"(ga)                                                       \
               : "memory")
#define WAIT_ASYNC() asm volatile("s_wait_asynccnt 0x0" ::: "memory")

__device__ __forceinline__ uint32_t lds_addr(const void* p) {
  return (uint32_t)(uintptr_t)p;  // LDS aperture: low 32 bits = LDS offset
}

// ---------------------------------------------------------------------------
// Elementwise / layout kernels
// ---------------------------------------------------------------------------
__global__ void k_zero(float* p, int n) {
  int i = blockIdx.x * blockDim.x + threadIdx.x;
  if (i < n) p[i] = 0.f;
}

// NCDHW fp32 -> NDHWC bf16
__global__ void k_nchw2ndhwc_bf16(const float* __restrict__ in, bf16* __restrict__ out,
                                  int C, int M) {
  int i = blockIdx.x * blockDim.x + threadIdx.x;
  if (i < C * M) {
    int c = i / M, m = i % M;
    out[m * C + c] = (bf16)in[i];
  }
}

// NDHWC fp32 -> NCDHW fp32 (for writing reference-layout outputs)
__global__ void k_ndhwc2nchw(const float* __restrict__ in, float* __restrict__ out,
                             int C, int M) {
  int i = blockIdx.x * blockDim.x + threadIdx.x;
  if (i < C * M) {
    int m = i / C, c = i % C;
    out[c * M + m] = in[i];
  }
}

__global__ void k_f32_to_bf16(const float* __restrict__ in, bf16* __restrict__ out, int n) {
  int i = blockIdx.x * blockDim.x + threadIdx.x;
  if (i < n) out[i] = (bf16)in[i];
}

// concat([a(Ca), b(Cb)], channel axis) in NDHWC, emitted as bf16
__global__ void k_concat_bf16(const float* __restrict__ a, int Ca,
                              const float* __restrict__ b, int Cb,
                              bf16* __restrict__ out, int M) {
  int C = Ca + Cb;
  int i = blockIdx.x * blockDim.x + threadIdx.x;
  if (i < M * C) {
    int m = i / C, c = i % C;
    float v = (c < Ca) ? a[m * Ca + c] : b[m * Cb + (c - Ca)];
    out[i] = (bf16)v;
  }
}

__global__ void k_copy1(const float* __restrict__ s, float* __restrict__ d) {
  if (blockIdx.x == 0 && threadIdx.x == 0) d[0] = s[0];
}

// Weight transform: torch [O][I][kvol] fp32 -> [O][Kp] bf16 (row-major over K,
// K index = ko*I + ci, zero-padded to Kp = ceil32(I*kvol)).
__global__ void k_wx(const float* __restrict__ w, bf16* __restrict__ wt,
                     int O, int I, int kvol, int Kp) {
  int i = blockIdx.x * blockDim.x + threadIdx.x;
  int n = O * Kp;
  if (i < n) {
    int o = i / Kp, k = i % Kp;
    float v = 0.f;
    if (k < I * kvol) {
      int ci = k % I;
      int ko = k / I;
      v = w[(o * I + ci) * kvol + ko];
    }
    wt[i] = (bf16)v;
  }
}

// ---------------------------------------------------------------------------
// Implicit-GEMM conv3d on WMMA (bf16 inputs, fp32 accumulate).
// One wave computes a 16(M voxels) x 16*NT(N=Cout) tile:
//  - tiles staged with global_load_async_to_lds_b128 (ASYNCcnt path),
//    zero-fill via plain LDS stores for OOB/padded rows,
//  - one A fragment reused across NT WMMAs (register blocking over N).
// ---------------------------------------------------------------------------
template <int NT>
__global__ __launch_bounds__(32) void k_conv_wmma(
    const bf16* __restrict__ in, const bf16* __restrict__ wtT, int Kp,
    const float* __restrict__ bias, const float* __restrict__ res,
    float* __restrict__ out,
    int iD, int iH, int iW, int oD, int oH, int oW,
    int Cin, int Cout, int ks, int stride, int pad) {
  __shared__ bf16 lA[16][32];
  __shared__ bf16 lBt[16 * NT][32];  // [n][k] (B transposed -> contiguous frag reads)
  const int lane = threadIdx.x;
  const int row = lane & 15;
  const int half = lane >> 4;  // 0 or 1
  const int mtile = blockIdx.x, ntile = blockIdx.y;
  const int M = oD * oH * oW;
  const int Ktot = ks * ks * ks * Cin;
  const int n0 = ntile * 16 * NT;
  v8f acc[NT];
#pragma unroll
  for (int nt = 0; nt < NT; ++nt) acc[nt] = (v8f){};

  // voxel coords for this lane's A-staging row
  const int gmA = mtile * 16 + row;
  const bool mvalid = gmA < M;
  const int gmc = mvalid ? gmA : 0;
  const int ox = gmc % oW;
  const int t2 = gmc / oW;
  const int oy = t2 % oH;
  const int oz = t2 / oH;
  const bool fastA = (Cin & 15) == 0;
  const uint4 zero4 = {0u, 0u, 0u, 0u};

  for (int k0 = 0; k0 < Kp; k0 += 32) {
    // --- stage B tile: (16*NT) rows x 32 K via async b128, rows >= Cout zero ---
    __builtin_prefetch((const void*)(wtT + (size_t)n0 * Kp + k0 + 64), 0, 3);
#pragma unroll
    for (int r = 0; r < NT * 2; ++r) {
      int q = r * 32 + lane;  // 16-byte slot in tile
      int n = q >> 2;
      int qq = q & 3;
      int gn = n0 + n;
      bf16* dst = &lBt[n][qq * 8];
      if (gn < Cout) {
        uint64_t ga = (uint64_t)(uintptr_t)(wtT + (size_t)gn * Kp + k0 + qq * 8);
        ASYNC_LD_B128(lds_addr(dst), ga);
      } else {
        *(uint4*)dst = zero4;
      }
    }
    // --- stage A tile: 16 voxels x 32 K ---
    if (fastA) {
      // chunk = two 16-element contiguous channel segments of one tap each
#pragma unroll
      for (int seg = 0; seg < 2; ++seg) {
        int k0s = k0 + seg * 16;
        int cin0 = k0s % Cin;  // uniform, multiple of 16
        int ko = k0s / Cin;
        int kx = ko % ks;
        int tt = ko / ks;
        int ky = tt % ks;
        int kz = tt / ks;
        int ix = ox * stride - pad + kx;
        int iy = oy * stride - pad + ky;
        int iz = oz * stride - pad + kz;
        bf16* dst = &lA[row][seg * 16 + half * 8];
        if (mvalid && (unsigned)ix < (unsigned)iW && (unsigned)iy < (unsigned)iH &&
            (unsigned)iz < (unsigned)iD) {
          uint64_t ga = (uint64_t)(uintptr_t)(
              in + ((iz * iH + iy) * iW + ix) * Cin + cin0 + half * 8);
          ASYNC_LD_B128(lds_addr(dst), ga);
        } else {
          *(uint4*)dst = zero4;
        }
      }
    } else {
      // generic scalar gather (Cin=4 paths: init conv, att conv 1)
#pragma unroll
      for (int j = 0; j < 16; ++j) {
        int f = lane * 16 + j;
        int m = f >> 5, kk = f & 31;
        float v = 0.f;
        int gm = mtile * 16 + m;
        int gk = k0 + kk;
        if (gm < M && gk < Ktot) {
          int cin = gk % Cin;
          int ko = gk / Cin;
          int kx = ko % ks;
          int t = ko / ks;
          int ky = t % ks;
          int kz = t / ks;
          int gx = gm % oW;
          int g2 = gm / oW;
          int gy = g2 % oH;
          int gz = g2 / oH;
          int ix = gx * stride - pad + kx;
          int iy = gy * stride - pad + ky;
          int iz = gz * stride - pad + kz;
          if ((unsigned)ix < (unsigned)iW && (unsigned)iy < (unsigned)iH &&
              (unsigned)iz < (unsigned)iD)
            v = (float)in[((iz * iH + iy) * iW + ix) * Cin + cin];
        }
        lA[m][kk] = (bf16)v;
      }
    }
    WAIT_ASYNC();     // async tile transfers complete (ASYNCcnt == 0)
    __syncthreads();  // zero-fill DS stores + cross-lane visibility

    // --- fragments (ISA wave32 16-bit layouts), contiguous b128 DS reads ---
    union { v16bf v; uint4 q[2]; } A;
    const int kbA = half * 8;  // lanes 0-15: K 0..7/16..23 ; 16-31: K 8..15/24..31
    A.q[0] = *(const uint4*)(&lA[row][kbA]);
    A.q[1] = *(const uint4*)(&lA[row][kbA + 16]);
    const int kbB = half * 8;  // lanes 0-15: K 0..7/16..23 ; 16-31: K 8..15/24..31
#pragma unroll
    for (int nt = 0; nt < NT; ++nt) {
      union { v16bf v; uint4 q[2]; } B;
      B.q[0] = *(const uint4*)(&lBt[nt * 16 + row][kbB]);
      B.q[1] = *(const uint4*)(&lBt[nt * 16 + row][kbB + 16]);
      acc[nt] = __builtin_amdgcn_wmma_f32_16x16x32_bf16(false, A.v, false, B.v,
                                                        (short)0, acc[nt], false,
                                                        false);
    }
    __syncthreads();  // protect LDS WAR vs next iteration's async writes
  }

  // --- store: C/D layout — VGPR i: lanes 0-15 M=i, lanes 16-31 M=i+8 ---
  const int mbase = mtile * 16 + half * 8;
#pragma unroll
  for (int nt = 0; nt < NT; ++nt) {
    int gn = n0 + nt * 16 + row;
    if (gn < Cout) {
      float bs = bias[gn];
#pragma unroll
      for (int i = 0; i < 8; ++i) {
        int gm = mbase + i;
        if (gm < M) {
          int idx = gm * Cout + gn;
          float v = acc[nt][i] + bs;
          if (res) v += res[idx];
          out[idx] = v;
        }
      }
    }
  }
}

// ---------------------------------------------------------------------------
// GroupNorm (8 groups) stats + apply(+ReLU) -> bf16
// ---------------------------------------------------------------------------
__global__ void k_gn_stats(const float* __restrict__ x, int M, int C, int cpg,
                           float* __restrict__ stats) {
  __shared__ float s1[256], s2[256];
  int g = blockIdx.y;
  int n = M * cpg;
  float a = 0.f, b = 0.f;
  for (int i = blockIdx.x * blockDim.x + threadIdx.x; i < n;
       i += gridDim.x * blockDim.x) {
    int m = i / cpg, c = g * cpg + i % cpg;
    float v = x[m * C + c];
    a += v;
    b += v * v;
  }
  s1[threadIdx.x] = a;
  s2[threadIdx.x] = b;
  __syncthreads();
  for (int s = 128; s > 0; s >>= 1) {
    if ((int)threadIdx.x < s) {
      s1[threadIdx.x] += s1[threadIdx.x + s];
      s2[threadIdx.x] += s2[threadIdx.x + s];
    }
    __syncthreads();
  }
  if (threadIdx.x == 0) {
    atomicAdd(&stats[g], s1[0]);
    atomicAdd(&stats[8 + g], s2[0]);
  }
}

__global__ void k_gn_apply(const float* __restrict__ x, const float* __restrict__ stats,
                           const float* __restrict__ gamma, const float* __restrict__ beta,
                           bf16* __restrict__ out, int M, int C, int cpg) {
  int i = blockIdx.x * blockDim.x + threadIdx.x;
  if (i < M * C) {
    int c = i % C;
    int g = c / cpg;
    float icnt = 1.f / ((float)M * (float)cpg);
    float mu = stats[g] * icnt;
    float var = stats[8 + g] * icnt - mu * mu;
    float v = (x[i] - mu) * rsqrtf(var + 1e-5f) * gamma[c] + beta[c];
    out[i] = (bf16)fmaxf(v, 0.f);  // GN always followed by ReLU in this net
  }
}

// ---------------------------------------------------------------------------
// ConvTranspose3d k=2 s=2 (no overlap)
// ---------------------------------------------------------------------------
__global__ void k_convT(const float* __restrict__ in, const float* __restrict__ w,
                        const float* __restrict__ bias, float* __restrict__ out,
                        int iD, int iH, int iW, int Cin, int Cout) {
  int oH = iH * 2, oW = iW * 2;
  int n = iD * 2 * oH * oW * Cout;
  int i = blockIdx.x * blockDim.x + threadIdx.x;
  if (i < n) {
    int nc = i % Cout;
    int m = i / Cout;
    int ox = m % oW;
    int t = m / oW;
    int oy = t % oH;
    int oz = t / oH;
    int ix = ox >> 1, px = ox & 1;
    int iy = oy >> 1, py = oy & 1;
    int iz = oz >> 1, pz = oz & 1;
    const float* ip = in + ((iz * iH + iy) * iW + ix) * Cin;
    int wbase = pz * 4 + py * 2 + px;
    float acc = bias[nc];
    for (int ci = 0; ci < Cin; ++ci) acc += ip[ci] * w[(ci * Cout + nc) * 8 + wbase];
    out[i] = acc;
  }
}

// ---------------------------------------------------------------------------
// "both": per-voxel argmax over 4 attention channels, category means, euclid
// ---------------------------------------------------------------------------
__global__ void k_both_acc(const float* __restrict__ x4, const float* __restrict__ att,
                           float* __restrict__ sums, float* __restrict__ cnt,
                           int* __restrict__ lbl, int M, int C) {
  int m = blockIdx.x * blockDim.x + threadIdx.x;
  if (m < M) {
    float best = att[m * 4];
    int k = 0;
    for (int j = 1; j < 4; ++j) {
      float v = att[m * 4 + j];
      if (v > best) { best = v; k = j; }
    }
    lbl[m] = k;
    atomicAdd(&cnt[k], 1.f);
    for (int c = 0; c < C; ++c) atomicAdd(&sums[c * 4 + k], x4[m * C + c]);
  }
}

__global__ void k_both_fin(const float* __restrict__ sums, const float* __restrict__ cnt,
                           float* __restrict__ avg, float* __restrict__ euc, int C) {
  int i = threadIdx.x;  // launch with 512 threads
  if (i < C * 4) {
    int k = i % 4;
    avg[i] = sums[i] / (cnt[k] + 1e-5f);
  }
  __syncthreads();
  if (i == 0) {
    float sumsq[4] = {0.f, 0.f, 0.f, 0.f};
    for (int k = 0; k < 4; ++k)
      for (int c = 0; c < C; ++c) {
        float v = avg[c * 4 + k];
        sumsq[k] += v * v;
      }
    float tot = 0.f;
    for (int k = 0; k < 4; ++k)
      for (int l = 0; l < 4; ++l) {
        float vec = 0.f;
        for (int c = 0; c < C; ++c) vec += avg[c * 4 + k] * avg[c * 4 + l];
        tot += 2.f * sumsq[k] - 2.f * vec;
      }
    euc[0] = sqrtf(tot);
  }
}

__global__ void k_both_apply(const float* __restrict__ x4, const float* __restrict__ avg,
                             const int* __restrict__ lbl, float* __restrict__ out,
                             int M, int C) {
  int i = blockIdx.x * blockDim.x + threadIdx.x;
  if (i < M * C) {
    int m = i / C, c = i % C;
    out[i] = x4[i] + avg[c * 4 + lbl[m]];
  }
}

// softmax over 4 channels, NDHWC in -> NCDHW out
__global__ void k_softmax4(const float* __restrict__ y, float* __restrict__ out, int M) {
  int m = blockIdx.x * blockDim.x + threadIdx.x;
  if (m < M) {
    float a = y[m * 4 + 0], b = y[m * 4 + 1], c = y[m * 4 + 2], d = y[m * 4 + 3];
    float mx = fmaxf(fmaxf(a, b), fmaxf(c, d));
    float ea = expf(a - mx), eb = expf(b - mx), ec = expf(c - mx), ed = expf(d - mx);
    float inv = 1.f / (ea + eb + ec + ed);
    out[0 * M + m] = ea * inv;
    out[1 * M + m] = eb * inv;
    out[2 * M + m] = ec * inv;
    out[3 * M + m] = ed * inv;
  }
}

// ---------------------------------------------------------------------------
// Host orchestration
// ---------------------------------------------------------------------------
struct W { bf16* p; int Kp; };
struct BlkW { W w1, w2; };

extern "C" void kernel_launch(void* const* d_in, const int* in_sizes, int n_in,
                              void* d_out, int out_size, void* d_ws, size_t ws_size,
                              hipStream_t stream) {
  (void)in_sizes; (void)n_in; (void)out_size; (void)ws_size;
  const int D0 = 96, D1 = 48, D2 = 24, D3 = 12;
  const int M0 = D0 * D0 * D0, M1 = D1 * D1 * D1, M2 = D2 * D2 * D2, M3 = D3 * D3 * D3;

  char* base = (char*)d_ws;
  size_t off = 0;
  auto alloc = [&](size_t bytes) -> void* {
    void* p = base + off;
    off += (bytes + 255) & ~(size_t)255;
    return p;
  };
  auto F = [&](int idx) -> const float* { return (const float*)d_in[idx]; };

  // ---- weight transforms (fp32 OIDHW -> bf16 [O][Kp], K zero-padded) ----
  auto XW = [&](int idx, int O, int I, int kvol) -> W {
    int Kp = ((I * kvol + 31) / 32) * 32;
    bf16* dst = (bf16*)alloc((size_t)O * Kp * sizeof(bf16));
    int n = O * Kp;
    k_wx<<<ceil_div(n, 256), 256, 0, stream>>>(F(idx), dst, O, I, kvol, Kp);
    return {dst, Kp};
  };
  auto XB = [&](int B, int C) -> BlkW { return {XW(B + 2, C, C, 27), XW(B + 6, C, C, 27)}; };

  W wInit = XW(1, 16, 4, 27);
  BlkW wEn1 = XB(3, 16);
  W wDn1 = XW(11, 32, 16, 27);
  BlkW wEn21 = XB(13, 32), wEn22 = XB(21, 32);
  W wDn2 = XW(29, 64, 32, 27);
  BlkW wEn31 = XB(31, 64), wEn32 = XB(39, 64);
  W wDn3 = XW(47, 128, 64, 27);
  BlkW wEn41 = XB(49, 128), wEn42 = XB(57, 128), wEn43 = XB(65, 128), wEn44 = XB(73, 128);
  W wAtt1 = XW(81, 16, 4, 27);
  W wAtt2 = XW(83, 16, 16, 27);
  W wAtt3 = XW(85, 4, 16, 27);
  W wD3c1 = XW(87, 64, 128, 1);
  W wD3c3 = XW(91, 64, 128, 1);
  BlkW wDe3 = XB(93, 64);
  W wD2c1 = XW(101, 32, 64, 1);
  W wD2c3 = XW(105, 32, 64, 1);
  BlkW wDe2 = XB(107, 32);
  W wD1c1 = XW(115, 16, 32, 1);
  W wD1c3 = XW(119, 16, 32, 1);
  BlkW wDe1 = XB(121, 16);
  W wEnd = XW(127, 4, 16, 1);

  // ---- activation buffers ----
  float* stats = (float*)alloc(16 * sizeof(float));
  bf16* x0b = (bf16*)alloc((size_t)M0 * 4 * sizeof(bf16));
  float* xinit = (float*)alloc((size_t)M0 * 16 * sizeof(float));
  float* x11 = (float*)alloc((size_t)M0 * 16 * sizeof(float));
  float* tmp0 = (float*)alloc((size_t)M0 * 16 * sizeof(float));
  bf16* bf0 = (bf16*)alloc((size_t)M0 * 32 * sizeof(bf16));
  float* x12 = (float*)alloc((size_t)M1 * 32 * sizeof(float));
  float* x2m = (float*)alloc((size_t)M1 * 32 * sizeof(float));
  float* x21 = (float*)alloc((size_t)M1 * 32 * sizeof(float));
  float* tmp1 = (float*)alloc((size_t)M1 * 32 * sizeof(float));
  bf16* bf1 = (bf16*)alloc((size_t)M1 * 64 * sizeof(bf16));
  float* x22 = (float*)alloc((size_t)M2 * 64 * sizeof(float));
  float* x3m = (float*)alloc((size_t)M2 * 64 * sizeof(float));
  float* x31 = (float*)alloc((size_t)M2 * 64 * sizeof(float));
  float* tmp2 = (float*)alloc((size_t)M2 * 64 * sizeof(float));
  bf16* bf2 = (bf16*)alloc((size_t)M2 * 128 * sizeof(bf16));
  float* x32 = (float*)alloc((size_t)M3 * 128 * sizeof(float));
  float* tmp3 = (float*)alloc((size_t)M3 * 128 * sizeof(float));
  bf16* bf3 = (bf16*)alloc((size_t)M3 * 128 * sizeof(bf16));
  float* x4post = (float*)alloc((size_t)M3 * 128 * sizeof(float));
  float* att1 = (float*)alloc((size_t)M1 * 16 * sizeof(float));
  float* att2 = (float*)alloc((size_t)M2 * 16 * sizeof(float));
  float* att3 = (float*)alloc((size_t)M3 * 4 * sizeof(float));
  float* bsums = (float*)alloc(516 * sizeof(float));  // sums[128*4] + cnt[4]
  float* bcnt = bsums + 512;
  float* bavg = (float*)alloc(512 * sizeof(float));
  int* lbl = (int*)alloc((size_t)M3 * sizeof(int));
  float* d3x1 = (float*)alloc((size_t)M3 * 64 * sizeof(float));
  float* d3y = (float*)alloc((size_t)M2 * 64 * sizeof(float));
  float* d3c = (float*)alloc((size_t)M2 * 64 * sizeof(float));
  float* y3 = (float*)alloc((size_t)M2 * 64 * sizeof(float));
  float* d2x1 = (float*)alloc((size_t)M2 * 32 * sizeof(float));
  float* d2y = (float*)alloc((size_t)M1 * 32 * sizeof(float));
  float* d2c = (float*)alloc((size_t)M1 * 32 * sizeof(float));
  float* y2 = (float*)alloc((size_t)M1 * 32 * sizeof(float));
  float* d1x1 = (float*)alloc((size_t)M1 * 16 * sizeof(float));
  float* d1y = (float*)alloc((size_t)M0 * 16 * sizeof(float));
  float* d1c = (float*)alloc((size_t)M0 * 16 * sizeof(float));
  float* y1 = (float*)alloc((size_t)M0 * 16 * sizeof(float));
  float* endo = (float*)alloc((size_t)M0 * 4 * sizeof(float));

  // ---- output slots (reference tuple order, flat) ----
  float* o_out = (float*)d_out;          // softmax [4][M0]
  float* o_att = o_out + 4 * M0;         // att     [4][M3]
  float* o_euc = o_att + 4 * M3;         // scalar
  float* o_wa = o_euc + 1;               // scalar
  float* o_x4pre = o_wa + 1;             // [128][M3]
  float* o_x4post = o_x4pre + 128 * M3;  // [128][M3]

  // ---- helpers ----
  auto CONV = [&](const bf16* in, W w, int bidx, const float* res, float* out,
                  int iDd, int oDd, int Cin, int Cout, int ks, int st, int pad) {
    int M = oDd * oDd * oDd;
    if (Cout >= 64) {
      dim3 g(ceil_div(M, 16), ceil_div(Cout, 64));
      k_conv_wmma<4><<<g, 32, 0, stream>>>(in, w.p, w.Kp, F(bidx), res, out, iDd,
                                           iDd, iDd, oDd, oDd, oDd, Cin, Cout, ks,
                                           st, pad);
    } else if (Cout == 32) {
      dim3 g(ceil_div(M, 16), 1);
      k_conv_wmma<2><<<g, 32, 0, stream>>>(in, w.p, w.Kp, F(bidx), res, out, iDd,
                                           iDd, iDd, oDd, oDd, oDd, Cin, Cout, ks,
                                           st, pad);
    } else {
      dim3 g(ceil_div(M, 16), 1);
      k_conv_wmma<1><<<g, 32, 0, stream>>>(in, w.p, w.Kp, F(bidx), res, out, iDd,
                                           iDd, iDd, oDd, oDd, oDd, Cin, Cout, ks,
                                           st, pad);
    }
  };
  auto GN = [&](const float* x, const float* gs, const float* gb, bf16* out, int M,
                int C) {
    k_zero<<<1, 32, 0, stream>>>(stats, 16);
    dim3 g(256, 8);
    k_gn_stats<<<g, 256, 0, stream>>>(x, M, C, C / 8, stats);
    int n = M * C;
    k_gn_apply<<<ceil_div(n, 256), 256, 0, stream>>>(x, stats, gs, gb, out, M, C,
                                                     C / 8);
  };
  auto RES = [&](int B, BlkW w, const float* xin, float* xout, float* tmp, bf16* bf,
                 int Dd, int C) {
    int M = Dd * Dd * Dd;
    GN(xin, F(B), F(B + 1), bf, M, C);
    CONV(bf, w.w1, B + 3, nullptr, tmp, Dd, Dd, C, C, 3, 1, 1);
    GN(tmp, F(B + 4), F(B + 5), bf, M, C);
    CONV(bf, w.w2, B + 7, xin, xout, Dd, Dd, C, C, 3, 1, 1);
  };
  auto TOBF = [&](const float* x, bf16* out, int n) {
    k_f32_to_bf16<<<ceil_div(n, 256), 256, 0, stream>>>(x, out, n);
  };

  // ---- encoder ----
  k_nchw2ndhwc_bf16<<<ceil_div(4 * M0, 256), 256, 0, stream>>>(F(0), x0b, 4, M0);
  CONV(x0b, wInit, 2, nullptr, xinit, D0, D0, 4, 16, 3, 1, 1);
  RES(3, wEn1, xinit, x11, tmp0, bf0, D0, 16);
  TOBF(x11, bf0, M0 * 16);
  CONV(bf0, wDn1, 12, nullptr, x12, D0, D1, 16, 32, 3, 2, 1);
  RES(13, wEn21, x12, x2m, tmp1, bf1, D1, 32);
  RES(21, wEn22, x2m, x21, tmp1, bf1, D1, 32);
  TOBF(x21, bf1, M1 * 32);
  CONV(bf1, wDn2, 30, nullptr, x22, D1, D2, 32, 64, 3, 2, 1);
  RES(31, wEn31, x22, x3m, tmp2, bf2, D2, 64);
  RES(39, wEn32, x3m, x31, tmp2, bf2, D2, 64);
  TOBF(x31, bf2, M2 * 64);
  CONV(bf2, wDn3, 48, nullptr, x32, D2, D3, 64, 128, 3, 2, 1);
  RES(49, wEn41, x32, x32, tmp3, bf3, D3, 128);
  RES(57, wEn42, x32, x32, tmp3, bf3, D3, 128);
  RES(65, wEn43, x32, x32, tmp3, bf3, D3, 128);
  RES(73, wEn44, x32, x32, tmp3, bf3, D3, 128);
  k_ndhwc2nchw<<<ceil_div(128 * M3, 256), 256, 0, stream>>>(x32, o_x4pre, 128, M3);

  // ---- attention branch ----
  CONV(x0b, wAtt1, 82, nullptr, att1, D0, D1, 4, 16, 3, 2, 1);
  TOBF(att1, bf1, M1 * 16);
  CONV(bf1, wAtt2, 84, nullptr, att2, D1, D2, 16, 16, 3, 2, 1);
  TOBF(att2, bf2, M2 * 16);
  CONV(bf2, wAtt3, 86, nullptr, att3, D2, D3, 16, 4, 3, 2, 1);
  k_ndhwc2nchw<<<ceil_div(4 * M3, 256), 256, 0, stream>>>(att3, o_att, 4, M3);

  // ---- both ----
  k_zero<<<ceil_div(516, 256), 256, 0, stream>>>(bsums, 516);
  k_both_acc<<<ceil_div(M3, 128), 128, 0, stream>>>(x32, att3, bsums, bcnt, lbl, M3, 128);
  k_both_fin<<<1, 512, 0, stream>>>(bsums, bcnt, bavg, o_euc, 128);
  k_both_apply<<<ceil_div(M3 * 128, 256), 256, 0, stream>>>(x32, bavg, lbl, x4post, M3, 128);
  k_ndhwc2nchw<<<ceil_div(128 * M3, 256), 256, 0, stream>>>(x4post, o_x4post, 128, M3);
  k_copy1<<<1, 32, 0, stream>>>(F(129), o_wa);

  // ---- decoder ----
  // deup3 + de3
  TOBF(x4post, bf3, M3 * 128);
  CONV(bf3, wD3c1, 88, nullptr, d3x1, D3, D3, 128, 64, 1, 1, 0);
  {
    int n = M2 * 64;
    k_convT<<<ceil_div(n, 256), 256, 0, stream>>>(d3x1, F(89), F(90), d3y, D3, D3, D3, 64, 64);
  }
  k_concat_bf16<<<ceil_div(M2 * 128, 256), 256, 0, stream>>>(x31, 64, d3y, 64, bf2, M2);
  CONV(bf2, wD3c3, 92, nullptr, d3c, D2, D2, 128, 64, 1, 1, 0);
  RES(93, wDe3, d3c, y3, tmp2, bf2, D2, 64);
  // deup2 + de2
  TOBF(y3, bf2, M2 * 64);
  CONV(bf2, wD2c1, 102, nullptr, d2x1, D2, D2, 64, 32, 1, 1, 0);
  {
    int n = M1 * 32;
    k_convT<<<ceil_div(n, 256), 256, 0, stream>>>(d2x1, F(103), F(104), d2y, D2, D2, D2, 32, 32);
  }
  k_concat_bf16<<<ceil_div(M1 * 64, 256), 256, 0, stream>>>(x21, 32, d2y, 32, bf1, M1);
  CONV(bf1, wD2c3, 106, nullptr, d2c, D1, D1, 64, 32, 1, 1, 0);
  RES(107, wDe2, d2c, y2, tmp1, bf1, D1, 32);
  // deup1 + de1
  TOBF(y2, bf1, M1 * 32);
  CONV(bf1, wD1c1, 116, nullptr, d1x1, D1, D1, 32, 16, 1, 1, 0);
  {
    int n = M0 * 16;
    k_convT<<<ceil_div(n, 256), 256, 0, stream>>>(d1x1, F(117), F(118), d1y, D1, D1, D1, 16, 16);
  }
  k_concat_bf16<<<ceil_div(M0 * 32, 256), 256, 0, stream>>>(x11, 16, d1y, 16, bf0, M0);
  CONV(bf0, wD1c3, 120, nullptr, d1c, D0, D0, 32, 16, 1, 1, 0);
  RES(121, wDe1, d1c, y1, tmp0, bf0, D0, 16);

  // ---- head ----
  TOBF(y1, bf0, M0 * 16);
  CONV(bf0, wEnd, 128, nullptr, endo, D0, D0, 16, 4, 1, 1, 0);
  k_softmax4<<<ceil_div(M0, 256), 256, 0, stream>>>(endo, o_out, M0);
}
